// VectorQuantizer_16879221473643
// MI455X (gfx1250) — matrix operand order repaired and checked by reading, not compiled, not run on hardware
//
#include <hip/hip_runtime.h>
#include <hip/hip_bf16.h>

typedef __attribute__((ext_vector_type(16))) _Float16 v16h;
typedef __attribute__((ext_vector_type(8)))  float    v8f;

#define VQ_NE    512
#define VQ_DIM   64
#define VQ_HW    4096                 // 64*64
#define VQ_NB    32
#define VQ_NVEC  (VQ_NB * VQ_HW)      // 131072 vectors
#define VQ_OUT   (VQ_NVEC * VQ_DIM)   // 8388608 output elements
#define VQ_NPART 32768                // blocks in gather/loss kernel

// ---------------------------------------------------------------------------
// Kernel 1: codebook -> f16 copy + per-row squared norms (f32)
// ---------------------------------------------------------------------------
__global__ __launch_bounds__(256) void vq_setup(const float* __restrict__ emb,
                                                _Float16* __restrict__ emb_h,
                                                float* __restrict__ normsq) {
  int row = blockIdx.x * 256 + threadIdx.x;
  if (row < VQ_NE) {
    const float* e = emb + row * VQ_DIM;
    float s = 0.0f;
#pragma unroll
    for (int c = 0; c < VQ_DIM; ++c) {
      float v = e[c];
      s += v * v;
      emb_h[row * VQ_DIM + c] = (_Float16)v;
    }
    normsq[row] = s;
  }
}

// ---------------------------------------------------------------------------
// Kernel 2: WMMA distance matmul + argmin.  One wave handles 16 z-vectors
// against all 512 codes.  score(M,N) = ||e_N||^2 + (-2*z_M) . e_N, injected
// via the C accumulator of v_wmma_f32_16x16x32_f16.
// ---------------------------------------------------------------------------
__global__ __launch_bounds__(256) void vq_argmin(const float* __restrict__ z,
                                                 const _Float16* __restrict__ emb_h,
                                                 const float* __restrict__ normsq,
                                                 int* __restrict__ idx_out) {
  const int lane = threadIdx.x & 31;
  const int wave = threadIdx.x >> 5;
  const int tile = blockIdx.x * 8 + wave;   // 0..8191 row tiles
  const int n0   = tile * 16;
  const int b    = n0 >> 12;                // 4096 vectors per batch image
  const int hwb  = n0 & 4095;
  const int l16  = lane & 15;
  const bool hi  = lane >= 16;

  // ---- A fragments (16x32 f16 layout).  Row M = l16, K = channel c.
  // z element (n, c) = z[(b*64 + c)*4096 + hw].  Half-wave lanes share c and
  // have consecutive hw -> 64B coalesced segments per channel.
  const float* zrow = z + (size_t)b * (VQ_DIM * VQ_HW) + (size_t)(hwb + l16);
  const int cb = hi ? 16 : 0;  // lanes 0-15: K 0..15 ; lanes 16-31: K 16..31
  v16h a0, a1;
#pragma unroll
  for (int i = 0; i < 8; ++i) {
    a0[i]     = (_Float16)(-2.0f * zrow[(size_t)(cb + i)      * VQ_HW]);
    a0[i + 8] = (_Float16)(-2.0f * zrow[(size_t)(cb + 8 + i)  * VQ_HW]);
    a1[i]     = (_Float16)(-2.0f * zrow[(size_t)(cb + 32 + i) * VQ_HW]);
    a1[i + 8] = (_Float16)(-2.0f * zrow[(size_t)(cb + 40 + i) * VQ_HW]);
  }

  float mv[8];
  int   mc[8];
#pragma unroll
  for (int r = 0; r < 8; ++r) { mv[r] = 3.0e38f; mc[r] = 0; }

  const int koff = hi ? 16 : 0;  // B layout: lanes 0-15 K=0..15, lanes 16-31 K=16..31
  for (int t = 0; t < VQ_NE / 16; ++t) {
    const int col = t * 16 + l16;                 // column N handled by this lane
    const _Float16* eb = emb_h + (size_t)col * VQ_DIM + koff;
    v16h b0 = *(const v16h*)(eb);                 // K 0..31 slice (32B aligned)
    v16h b1 = *(const v16h*)(eb + 32);            // K 32..63 slice
    float nrm = normsq[col];
    v8f c8;
#pragma unroll
    for (int r = 0; r < 8; ++r) c8[r] = nrm;      // C(M,N) = ||e_N||^2
    c8 = __builtin_amdgcn_wmma_f32_16x16x32_f16(false, a0, false, b0,
                                                (short)0, c8, false, false);
    c8 = __builtin_amdgcn_wmma_f32_16x16x32_f16(false, a1, false, b1,
                                                (short)0, c8, false, false);
#pragma unroll
    for (int r = 0; r < 8; ++r) {
      float s = c8[r];
      bool p = s < mv[r];
      mv[r] = p ? s   : mv[r];
      mc[r] = p ? col : mc[r];
    }
  }

  // ---- argmin across the 16 lanes of each half (D layout: VGPR r holds
  // row M=r on lanes 0-15 and row M=r+8 on lanes 16-31).
#pragma unroll
  for (int off = 1; off < 16; off <<= 1) {
#pragma unroll
    for (int r = 0; r < 8; ++r) {
      float ov = __shfl_xor(mv[r], off, 32);
      int   oc = __shfl_xor(mc[r], off, 32);
      bool p = (ov < mv[r]) || (ov == mv[r] && oc < mc[r]);  // first-index tiebreak
      mv[r] = p ? ov : mv[r];
      mc[r] = p ? oc : mc[r];
    }
  }
  if (lane == 0) {
#pragma unroll
    for (int r = 0; r < 8; ++r) idx_out[n0 + r] = mc[r];
  } else if (lane == 16) {
#pragma unroll
    for (int r = 0; r < 8; ++r) idx_out[n0 + 8 + r] = mc[r];
  }
}

// ---------------------------------------------------------------------------
// Kernel 3: gather z_q (exact f32 codebook rows) into [B,C,H,W] output and
// reduce sum((z_q - z)^2) to one partial per block (deterministic).
// ---------------------------------------------------------------------------
__global__ __launch_bounds__(256) void vq_gather_loss(const float* __restrict__ z,
                                                      const float* __restrict__ emb,
                                                      const int* __restrict__ idx,
                                                      float* __restrict__ out,     // d_out + 1
                                                      float* __restrict__ partial) {
  __shared__ float red[8];
  size_t o = (size_t)blockIdx.x * 256 + threadIdx.x;   // [B,C,H,W] flat index
  int hw = (int)(o & 4095);
  int c  = (int)((o >> 12) & 63);
  int b  = (int)(o >> 18);
  int n  = b * VQ_HW + hw;
  float q  = emb[idx[n] * VQ_DIM + c];
  float zv = z[o];
  out[o] = q;                                          // straight-through fwd == z_q
  float d = q - zv;
  float s = d * d;
#pragma unroll
  for (int off = 16; off >= 1; off >>= 1) s += __shfl_xor(s, off, 32);
  int lane = threadIdx.x & 31, wv = threadIdx.x >> 5;
  if (lane == 0) red[wv] = s;
  __syncthreads();
  if (threadIdx.x == 0) {
    float t = 0.0f;
#pragma unroll
    for (int w = 0; w < 8; ++w) t += red[w];
    partial[blockIdx.x] = t;
  }
}

// ---------------------------------------------------------------------------
// Kernel 4: deterministic fixed-order reduction of partials -> loss scalar.
// ---------------------------------------------------------------------------
__global__ __launch_bounds__(256) void vq_final(const float* __restrict__ partial,
                                                float* __restrict__ loss) {
  __shared__ float red[256];
  float s = 0.0f;
  for (int k = 0; k < VQ_NPART / 256; ++k) s += partial[threadIdx.x + 256 * k];
  red[threadIdx.x] = s;
  __syncthreads();
  for (int st = 128; st >= 1; st >>= 1) {
    if (threadIdx.x < st) red[threadIdx.x] += red[threadIdx.x + st];
    __syncthreads();
  }
  if (threadIdx.x == 0) loss[0] = red[0] * (1.25f / (float)VQ_OUT);
}

// ---------------------------------------------------------------------------
extern "C" void kernel_launch(void* const* d_in, const int* in_sizes, int n_in,
                              void* d_out, int out_size, void* d_ws, size_t ws_size,
                              hipStream_t stream) {
  (void)in_sizes; (void)n_in; (void)out_size; (void)ws_size;
  const float* z   = (const float*)d_in[0];   // [32,64,64,64] f32
  const float* emb = (const float*)d_in[1];   // [512,64] f32
  float* out = (float*)d_out;                 // [0]=loss, [1..]=z_q in BCHW

  char* ws = (char*)d_ws;
  _Float16* emb_h  = (_Float16*)ws;                               //  65536 B
  float*    normsq = (float*)(ws + 65536);                        //   2048 B
  int*      idx    = (int*)(ws + 65536 + 2048);                   // 524288 B
  float*    part   = (float*)(ws + 65536 + 2048 + 524288);        // 131072 B

  vq_setup<<<2, 256, 0, stream>>>(emb, emb_h, normsq);
  vq_argmin<<<VQ_NVEC / (16 * 8), 256, 0, stream>>>(z, emb_h, normsq, idx);
  vq_gather_loss<<<VQ_NPART, 256, 0, stream>>>(z, emb, idx, out + 1, part);
  vq_final<<<1, 256, 0, stream>>>(part, out);
}
